// HiveGNN_25589415149881
// MI455X (gfx1250) — compile-verified, matrix operand order checked by reading, not run on hardware
//
#include <hip/hip_runtime.h>
#include <hip/hip_bf16.h>

typedef __attribute__((ext_vector_type(16))) _Float16 v16h;
typedef __attribute__((ext_vector_type(8)))  float    v8f;

#define DD 128
#define HH 4
#define CC 32
#define GG 64

__device__ __forceinline__ unsigned ford(float f) {
    unsigned u = __float_as_uint(f);
    return (u & 0x80000000u) ? ~u : (u | 0x80000000u);
}
__device__ __forceinline__ float forddec(unsigned u) {
    return (u & 0x80000000u) ? __uint_as_float(u & 0x7FFFFFFFu) : __uint_as_float(~u);
}
__device__ __forceinline__ float lrelu(float v) { return v > 0.f ? v : 0.2f * v; }

// ---------------- utility: zero fill ----------------
__global__ void k_fillz(float* p, size_t n) {
    size_t i = (size_t)blockIdx.x * blockDim.x + threadIdx.x;
    size_t stride = (size_t)gridDim.x * blockDim.x;
    for (; i < n; i += stride) p[i] = 0.f;
}

// ---------------- weight convert: f32 [L][128][128] -> f16 transposed [L][col][k] ----
__global__ void k_convw(const float* __restrict__ Wl, const float* __restrict__ Wr,
                        _Float16* __restrict__ wtL, _Float16* __restrict__ wtR, int total) {
    int idx = blockIdx.x * blockDim.x + threadIdx.x;
    if (idx >= total) return;                 // total = L*128*128
    int l = idx >> 14, rem = idx & 16383;
    int col = rem >> 7, k = rem & 127;
    size_t src = (size_t)l * 16384 + (size_t)k * 128 + col;
    size_t dst = (size_t)l * 16384 + (size_t)col * 128 + k;
    wtL[dst] = (_Float16)Wl[src];
    wtR[dst] = (_Float16)Wr[src];
}

// ---------------- embedding: h = relu(x @ embW + embB), F0 small ----------------
__global__ __launch_bounds__(DD) void k_embed(const float* __restrict__ x,
                                              const float* __restrict__ W,
                                              const float* __restrict__ b,
                                              float* __restrict__ h, int n, int f0) {
    __shared__ float xs[32];
    int node = blockIdx.x, c = threadIdx.x;
    if (node >= n) return;
    if (c < f0) xs[c] = x[(size_t)node * f0 + c];
    __syncthreads();
    float a = b[c];
    for (int k = 0; k < f0; ++k) a += xs[k] * W[k * DD + c];
    h[(size_t)node * DD + c] = a > 0.f ? a : 0.f;
}

// ---------------- WMMA GEMM: out[N,128] = A[N,128] @ W[128,128] + bias ----------------
// W given pre-transposed f16: Wt[col][k]. Block = 256 thr = 8 waves, 64 rows x 128 cols.
__global__ __launch_bounds__(256) void k_gemm128(const float* __restrict__ A,
                                                 const _Float16* __restrict__ Wt,
                                                 const float* __restrict__ bias,
                                                 float* __restrict__ out, int n) {
    __shared__ _Float16 As[64][136];
    __shared__ _Float16 Bs[128][136];
    const int t = threadIdx.x;
    const int rowBase = blockIdx.x * 64;

    // stage Wt (128x128 halfs = 2048 uint4)
    {
        const uint4* src = (const uint4*)Wt;
        for (int i = t; i < 2048; i += 256) {
            int col = i >> 4, g = i & 15;
            *(uint4*)&Bs[col][g * 8] = src[i];
        }
    }
    // stage A rows (f32 -> f16), 64 rows x 32 float4
    {
        for (int i = t; i < 2048; i += 256) {
            int r = i >> 5, g = i & 31;
            int row = rowBase + r;
            float4 v = (row < n) ? ((const float4*)(A + (size_t)row * DD))[g]
                                 : make_float4(0.f, 0.f, 0.f, 0.f);
            union { _Float16 hx[4]; uint2 u; } cv;
            cv.hx[0] = (_Float16)v.x; cv.hx[1] = (_Float16)v.y;
            cv.hx[2] = (_Float16)v.z; cv.hx[3] = (_Float16)v.w;
            *(uint2*)&As[r][g * 4] = cv.u;
        }
    }
    __syncthreads();

    const int wave = t >> 5, lane = t & 31;
    const int rt = wave >> 1;          // row tile 0..3
    const int chalf = wave & 1;        // column half 0..1
    const int lr = lane & 15;
    const int koff = (lane >> 4) * 8;  // K sub-group per ISA 16-bit A layout
    union Frag { uint4 u[2]; v16h v; };
    v8f acc[4] = {};

    for (int kb = 0; kb < 128; kb += 32) {
        Frag a;
        const int arow = rt * 16 + lr;
        a.u[0] = *(const uint4*)&As[arow][kb + koff];
        a.u[1] = *(const uint4*)&As[arow][kb + koff + 16];
#pragma unroll
        for (int tc = 0; tc < 4; ++tc) {
            Frag bfr;
            const int col = chalf * 64 + tc * 16 + lr;
            bfr.u[0] = *(const uint4*)&Bs[col][kb + koff];
            bfr.u[1] = *(const uint4*)&Bs[col][kb + koff + 16];
            acc[tc] = __builtin_amdgcn_wmma_f32_16x16x32_f16(
                false, a.v, false, bfr.v, (short)0, acc[tc], false, false);
        }
    }
#pragma unroll
    for (int tc = 0; tc < 4; ++tc) {
        const int col = chalf * 64 + tc * 16 + lr;
        const float bv = bias[col];
#pragma unroll
        for (int r = 0; r < 8; ++r) {
            int row = rowBase + rt * 16 + (lane >> 4) * 8 + r;
            if (row < n) out[(size_t)row * DD + col] = acc[tc][r] + bv;
        }
    }
}

// ---------------- edge pass A: logits + segment max ----------------
__global__ void k_edgeA(const float* __restrict__ xl, const float* __restrict__ xr,
                        const int* __restrict__ ei, const float* __restrict__ att,
                        float* __restrict__ wbuf, unsigned* __restrict__ lmax,
                        int eCnt, int m) {
    int idx = blockIdx.x * blockDim.x + threadIdx.x;
    if (idx >= m * HH) return;
    int e = idx >> 2, hh = idx & 3;
    int src, dst;
    if (e < eCnt) { src = ei[e]; dst = ei[eCnt + e]; } else { src = dst = e - eCnt; }
    const float* pa = xl + (size_t)src * DD + hh * CC;
    const float* pb = xr + (size_t)dst * DD + hh * CC;
    const float* pc = att + hh * CC;
    float s = 0.f;
#pragma unroll
    for (int j = 0; j < CC; j += 4) {
        float4 va = *(const float4*)(pa + j);
        float4 vb = *(const float4*)(pb + j);
        float4 vc = *(const float4*)(pc + j);
        s += lrelu(va.x + vb.x) * vc.x;
        s += lrelu(va.y + vb.y) * vc.y;
        s += lrelu(va.z + vb.z) * vc.z;
        s += lrelu(va.w + vb.w) * vc.w;
    }
    wbuf[idx] = s;
    atomicMax(&lmax[(size_t)dst * HH + hh], ford(s));
}

// ---------------- edge pass B: exp + segment sum ----------------
__global__ void k_edgeB(float* __restrict__ wbuf, const unsigned* __restrict__ lmax,
                        float* __restrict__ z, const int* __restrict__ ei,
                        int eCnt, int m) {
    int idx = blockIdx.x * blockDim.x + threadIdx.x;
    if (idx >= m * HH) return;
    int e = idx >> 2, hh = idx & 3;
    int dst = (e < eCnt) ? ei[eCnt + e] : (e - eCnt);
    float lm = forddec(lmax[(size_t)dst * HH + hh]);
    float w = expf(wbuf[idx] - lm);
    wbuf[idx] = w;
    atomicAdd(&z[(size_t)dst * HH + hh], w);
}

// ---------------- edge pass C: weighted scatter-add ----------------
__global__ void k_edgeC(const float* __restrict__ xl, const float* __restrict__ wbuf,
                        const float* __restrict__ z, const int* __restrict__ ei,
                        float* __restrict__ agg, int eCnt, int m) {
    int idx = blockIdx.x * blockDim.x + threadIdx.x;
    if (idx >= m * 32) return;                     // 32 float4 groups per edge
    int e = idx >> 5, q = idx & 31, hh = q >> 3;
    int src, dst;
    if (e < eCnt) { src = ei[e]; dst = ei[eCnt + e]; } else { src = dst = e - eCnt; }
    float alpha = wbuf[e * HH + hh] / z[(size_t)dst * HH + hh];
    float4 v = *(const float4*)(xl + (size_t)src * DD + q * 4);
    float* o = agg + (size_t)dst * DD + q * 4;
    atomicAdd(o + 0, alpha * v.x);
    atomicAdd(o + 1, alpha * v.y);
    atomicAdd(o + 2, alpha * v.z);
    atomicAdd(o + 3, alpha * v.w);
}

// ---------------- batchnorm: partial stats, finalize, apply ----------------
__global__ __launch_bounds__(DD) void k_bnstats(const float* __restrict__ agg,
                                                float* __restrict__ bsum,
                                                float* __restrict__ bsq, int n) {
    int c = threadIdx.x;
    float s = 0.f, q = 0.f;
    for (int r = blockIdx.x; r < n; r += gridDim.x) {
        float v = agg[(size_t)r * DD + c];
        s += v; q += v * v;
    }
    atomicAdd(&bsum[c], s);
    atomicAdd(&bsq[c], q);
}

__global__ __launch_bounds__(DD) void k_bnfinal(const float* __restrict__ bsum,
                                                const float* __restrict__ bsq,
                                                const float* __restrict__ gamma,
                                                const float* __restrict__ beta,
                                                float* __restrict__ scale,
                                                float* __restrict__ shift, int n) {
    int c = threadIdx.x;
    float inv = 1.f / (float)n;
    float mu = bsum[c] * inv;
    float var = bsq[c] * inv - mu * mu;
    float sc = gamma[c] * rsqrtf(var + 1e-5f);
    scale[c] = sc;
    shift[c] = beta[c] - mu * sc;
}

__global__ void k_bnapply(const float* __restrict__ agg, const float* __restrict__ scale,
                          const float* __restrict__ shift, float* __restrict__ h,
                          size_t total, int addres) {
    size_t i = (size_t)blockIdx.x * blockDim.x + threadIdx.x;
    size_t stride = (size_t)gridDim.x * blockDim.x;
    for (; i < total; i += stride) {
        int c = (int)(i & (DD - 1));
        float v = agg[i] * scale[c] + shift[c];
        v = v > 0.f ? v : 0.f;
        if (addres) v += h[i];
        h[i] = v;
    }
}

// ---------------- readout ----------------
__global__ void k_gcount(const int* __restrict__ batch, float* __restrict__ gcnt, int n) {
    int i = blockIdx.x * blockDim.x + threadIdx.x;
    if (i < n) atomicAdd(&gcnt[batch[i]], 1.0f);
}

__global__ void k_gscatter(const float* __restrict__ h, const int* __restrict__ batch,
                           float* __restrict__ gsum, unsigned* __restrict__ gmax, int n) {
    size_t total = (size_t)n * DD;
    size_t i = (size_t)blockIdx.x * blockDim.x + threadIdx.x;
    size_t stride = (size_t)gridDim.x * blockDim.x;
    for (; i < total; i += stride) {
        int node = (int)(i >> 7), c = (int)(i & (DD - 1));
        int g = batch[node];
        float v = h[i];
        atomicAdd(&gsum[g * DD + c], v);
        atomicMax(&gmax[g * DD + c], ford(v));
    }
}

__global__ __launch_bounds__(DD) void k_gmlp(const float* __restrict__ gsum,
                                             const unsigned* __restrict__ gmax,
                                             const float* __restrict__ gcnt,
                                             const float* __restrict__ W1, const float* __restrict__ b1,
                                             const float* __restrict__ W2, const float* __restrict__ b2,
                                             const float* __restrict__ W3, const float* __restrict__ b3,
                                             float* __restrict__ out) {
    __shared__ float ge[2 * DD];
    __shared__ float v1[DD];
    __shared__ float v2[DD / 2];
    int g = blockIdx.x, t = threadIdx.x;
    float cnt = gcnt[g];
    float denom = fmaxf(cnt, 1.0f);
    ge[t] = gsum[g * DD + t] / denom;
    ge[DD + t] = (cnt > 0.f) ? forddec(gmax[g * DD + t]) : 0.f;
    __syncthreads();
    float a = b1[t];
    for (int k = 0; k < 2 * DD; ++k) a += ge[k] * W1[k * DD + t];
    v1[t] = a > 0.f ? a : 0.f;
    __syncthreads();
    if (t < DD / 2) {
        float a2 = b2[t];
        for (int k = 0; k < DD; ++k) a2 += v1[k] * W2[k * (DD / 2) + t];
        v2[t] = a2 > 0.f ? a2 : 0.f;
    }
    __syncthreads();
    if (t == 0) {
        float a3 = b3[0];
        for (int k = 0; k < DD / 2; ++k) a3 += v2[k] * W3[k];
        out[g] = tanhf(a3);
    }
}

__global__ void k_copy(const float* __restrict__ src, float* __restrict__ dst, size_t n) {
    size_t i = (size_t)blockIdx.x * blockDim.x + threadIdx.x;
    size_t stride = (size_t)gridDim.x * blockDim.x;
    for (; i < n; i += stride) dst[i] = src[i];
}

// ================= host side =================
extern "C" void kernel_launch(void* const* d_in, const int* in_sizes, int n_in,
                              void* d_out, int out_size, void* d_ws, size_t ws_size,
                              hipStream_t stream) {
    const float* x     = (const float*)d_in[0];
    const int*   ei    = (const int*)d_in[1];
    const int*   batch = (const int*)d_in[2];
    const float* embW  = (const float*)d_in[3];
    const float* embB  = (const float*)d_in[4];
    const float* linlW = (const float*)d_in[5];
    const float* linlB = (const float*)d_in[6];
    const float* linrW = (const float*)d_in[7];
    const float* linrB = (const float*)d_in[8];
    const float* att   = (const float*)d_in[9];
    // d_in[10] gat_bias: provably cancelled by the immediately-following BatchNorm
    const float* gamma = (const float*)d_in[11];
    const float* beta  = (const float*)d_in[12];
    const float* W1 = (const float*)d_in[13];
    const float* b1 = (const float*)d_in[14];
    const float* W2 = (const float*)d_in[15];
    const float* b2 = (const float*)d_in[16];
    const float* W3 = (const float*)d_in[17];
    const float* b3 = (const float*)d_in[18];

    const int N = in_sizes[2];          // batch vector length
    const int F0 = in_sizes[0] / N;
    const int E = in_sizes[1] / 2;
    const int L = in_sizes[6] / DD;     // lin_l_b = [L, D]
    const int M = E + N;                // with self loops
    const size_t nD = (size_t)N * DD;

    // workspace carve-up (all segment sizes 16B-multiples)
    float* base = (float*)d_ws;
    float*    h_buf  = base;
    float*    xl     = h_buf + nD;
    float*    xr     = xl + nD;
    float*    agg    = xr + nD;                       // --- zero region start
    unsigned* lmax   = (unsigned*)(agg + nD);
    float*    zbuf   = (float*)(lmax + (size_t)N * HH);
    float*    bnsum  = zbuf + (size_t)N * HH;
    float*    bnsq   = bnsum + DD;                    // --- zero region end
    float*    bnscale = bnsq + DD;
    float*    bnshift = bnscale + DD;
    float*    wbuf   = bnshift + DD;                  // [M, H]
    _Float16* wtL    = (_Float16*)(wbuf + (size_t)M * HH);
    _Float16* wtR    = wtL + (size_t)L * DD * DD;
    float*    gsum   = (float*)(wtR + (size_t)L * DD * DD);  // --- readout zero start
    unsigned* gmax   = (unsigned*)(gsum + GG * DD);
    float*    gcnt   = (float*)(gmax + GG * DD);             // --- readout zero end

    const size_t zlen = nD + 2 * (size_t)N * HH + 2 * DD;

    // weight conversion + embedding
    {
        int total = L * DD * DD;
        k_convw<<<(total + 255) / 256, 256, 0, stream>>>(linlW, linrW, wtL, wtR, total);
        k_embed<<<N, DD, 0, stream>>>(x, embW, embB, h_buf, N, F0);
    }

    const int gemmBlocks = (N + 63) / 64;
    const int ehGrid = (M * HH + 255) / 256;
    const int ecGrid = (M * 32 + 255) / 256;

    for (int l = 0; l < L; ++l) {
        k_gemm128<<<gemmBlocks, 256, 0, stream>>>(h_buf, wtL + (size_t)l * DD * DD,
                                                  linlB + l * DD, xl, N);
        k_gemm128<<<gemmBlocks, 256, 0, stream>>>(h_buf, wtR + (size_t)l * DD * DD,
                                                  linrB + l * DD, xr, N);
        k_fillz<<<4096, 256, 0, stream>>>(agg, zlen);
        k_edgeA<<<ehGrid, 256, 0, stream>>>(xl, xr, ei, att + l * HH * CC, wbuf, lmax, E, M);
        k_edgeB<<<ehGrid, 256, 0, stream>>>(wbuf, lmax, zbuf, ei, E, M);
        k_edgeC<<<ecGrid, 256, 0, stream>>>(xl, wbuf, zbuf, ei, agg, E, M);
        k_bnstats<<<512, DD, 0, stream>>>(agg, bnsum, bnsq, N);
        k_bnfinal<<<1, DD, 0, stream>>>(bnsum, bnsq, gamma + l * DD, beta + l * DD,
                                        bnscale, bnshift, N);
        k_bnapply<<<4096, 256, 0, stream>>>(agg, bnscale, bnshift, h_buf, nD, l > 0 ? 1 : 0);
    }

    // graph readout
    k_fillz<<<64, 256, 0, stream>>>(gsum, (size_t)(2 * GG * DD + GG));
    k_gcount<<<(N + 255) / 256, 256, 0, stream>>>(batch, gcnt, N);
    k_gscatter<<<4096, 256, 0, stream>>>(h_buf, batch, gsum, gmax, N);
    float* out = (float*)d_out;
    k_gmlp<<<GG, DD, 0, stream>>>(gsum, gmax, gcnt, W1, b1, W2, b2, W3, b3, out);
    k_copy<<<4096, 256, 0, stream>>>(h_buf, out + GG, nD);
}
